// DualArchitectureRadialHashDisciplined_360777253471
// MI455X (gfx1250) — matrix operand order repaired and checked by reading, not compiled, not run on hardware
//
#include <hip/hip_runtime.h>
#include <math.h>

// ---------------------------------------------------------------------------
// Types for CDNA5 WMMA (wave32): 16x16x32 bf16 -> f32
// ---------------------------------------------------------------------------
typedef __bf16 bf16_t;
typedef __attribute__((ext_vector_type(16))) __bf16 v16bf;
typedef __attribute__((ext_vector_type(8)))  __bf16 v8bf;
typedef __attribute__((ext_vector_type(8)))  float  v8f;

// Native conversion: lets the backend use gfx1250 bf16 cvt ops.
__device__ __forceinline__ bf16_t f2bf(float f) { return (bf16_t)f; }

__device__ __forceinline__ v8f wmma_bf16(const v16bf& a, const v16bf& b, const v8f& c) {
    return __builtin_amdgcn_wmma_f32_16x16x32_bf16(false, a, false, b, (short)0, c,
                                                   false, false);
}

// ---------------------------------------------------------------------------
// BitNet scale: scale = max(mean(|W|), 1e-5), one block per matrix
// ---------------------------------------------------------------------------
__global__ void absmean_kernel(const float* __restrict__ w, int n, float* __restrict__ out) {
    __shared__ float red[256];
    float s = 0.f;
    for (int i = threadIdx.x; i < n; i += 256) s += fabsf(w[i]);
    red[threadIdx.x] = s;
    __syncthreads();
    for (int o = 128; o > 0; o >>= 1) {
        if ((int)threadIdx.x < o) red[threadIdx.x] += red[threadIdx.x + o];
        __syncthreads();
    }
    if (threadIdx.x == 0) out[0] = fmaxf(red[0] / (float)n, 1e-5f);
}

// ---------------------------------------------------------------------------
// Embedding: tok_emb[id] + radial_gain*(rt @ Wr^T) + hash_gain*hash_emb[bigram]
// One block (448 threads) per token.
// ---------------------------------------------------------------------------
__global__ void embed_kernel(const int* __restrict__ ids,
                             const float* __restrict__ tok_emb,
                             const float* __restrict__ rproj,
                             const float* __restrict__ rgain_p,
                             const float* __restrict__ hemb,
                             const float* __restrict__ hgain_p,
                             float* __restrict__ x, int T) {
    const int tok = blockIdx.x;            // 0..B*T-1
    const int t   = tok % T;
    const int d   = threadIdx.x;           // 0..447
    __shared__ float rf[4];
    __shared__ int   sbh;
    if (d == 0) {
        const int id = ids[tok];
        float re = 0.f, im = 0.f;
        float radius = 0.02f;                                // phi^0 * alpha
        const float phi = 1.6180339887498949f;
        for (int j = 0; j < 10; ++j) {
            float bit = (float)((id >> j) & 1);
            float ang = 0.6283185307179586f * (float)j;      // 2*pi/10 * j
            re += bit * radius * cosf(ang);
            im += bit * radius * sinf(ang);
            radius *= phi;
        }
        rf[0] = re;
        rf[1] = im;
        rf[2] = sqrtf(re * re + im * im + 1e-8f);
        rf[3] = atan2f(im, re + 1e-8f);
        const int prev = (t == 0) ? 0 : ids[tok - 1];
        sbh = (prev * 131 + id) & 1023;                      // % HASH_BUCKETS
    }
    __syncthreads();
    const int id = ids[tok];
    float v = tok_emb[(size_t)id * 448 + d];
    v += rgain_p[0] * (rf[0] * rproj[d * 4 + 0] + rf[1] * rproj[d * 4 + 1] +
                       rf[2] * rproj[d * 4 + 2] + rf[3] * rproj[d * 4 + 3]);
    v += hgain_p[0] * hemb[(size_t)sbh * 448 + d];
    x[(size_t)tok * 448 + d] = v;
}

// ---------------------------------------------------------------------------
// RMSNorm (f32 in, f32 out). One block per token.
// ---------------------------------------------------------------------------
__global__ void rmsnorm_kernel(const float* __restrict__ x, const float* __restrict__ w,
                               float* __restrict__ out, int C) {
    const int tok = blockIdx.x;
    __shared__ float red[256];
    float s = 0.f;
    for (int i = threadIdx.x; i < C; i += 256) {
        float v = x[(size_t)tok * C + i];
        s += v * v;
    }
    red[threadIdx.x] = s;
    __syncthreads();
    for (int o = 128; o > 0; o >>= 1) {
        if ((int)threadIdx.x < o) red[threadIdx.x] += red[threadIdx.x + o];
        __syncthreads();
    }
    const float rinv = rsqrtf(red[0] / (float)C + 1e-6f);
    for (int i = threadIdx.x; i < C; i += 256)
        out[(size_t)tok * C + i] = w[i] * x[(size_t)tok * C + i] * rinv;
}

// ---------------------------------------------------------------------------
// Tiled WMMA GEMM:  out[M,N] (+epilogue) = act[M,K] * quant(W[N,K])^T
//   - 128 threads = 4 waves, block tile 64x64, K-step 32
//   - double-buffered LDS (ping-pong), one barrier per K-step
//   - global_prefetch for the k+2 tile
//   - branchless BitNet quant (sign(w)*scale) folded into LDS staging
//   - epilogue: v = acc; if(gelu) gelu(v); v *= beta; if(resid) v += resid
// Requires M%64==0, N%64==0, K%32==0 (true for all shapes here).
// ---------------------------------------------------------------------------
__global__ void wmma_gemm_kernel(const float* __restrict__ act,
                                 const float* __restrict__ W,
                                 const float* __restrict__ scale_ptr,
                                 const float* __restrict__ resid,
                                 float beta, int do_gelu,
                                 float* __restrict__ out,
                                 int M, int N, int K) {
    __shared__ bf16_t As[2][64][32];
    __shared__ bf16_t Bs[2][64][32];

    const int tid  = threadIdx.x;
    const int lane = tid & 31;
    const int wave = tid >> 5;
    const int wm = wave & 1, wn = wave >> 1;          // 2x2 wave grid
    const int m0 = blockIdx.y * 64;
    const int n0 = blockIdx.x * 64;

    const bool doq = (scale_ptr != nullptr);
    const float qs = doq ? *scale_ptr : 0.f;

    v8f acc[2][2] = {};

    const int row  = tid >> 1;           // 0..63  (tile row loaded by this thread)
    const int colb = (tid & 1) * 16;     // 0 or 16 (16 consecutive k per thread)

    const int fr   = lane & 15;          // fragment row/col index
    const int kb8  = (lane >> 4) * 8;    // A-fragment k base
    const int kb16 = (lane >> 4) * 16;   // unused: B fragments also use kb8

    const float* aRow = act + (size_t)(m0 + row) * K + colb;
    const float* bRow = W   + (size_t)(n0 + row) * K + colb;

    alignas(16) float aReg[16];
    alignas(16) float bReg[16];

    auto gload = [&](int k0) {
        const float* sa = aRow + k0;
        *(float4*)(aReg + 0)  = *(const float4*)(sa + 0);
        *(float4*)(aReg + 4)  = *(const float4*)(sa + 4);
        *(float4*)(aReg + 8)  = *(const float4*)(sa + 8);
        *(float4*)(aReg + 12) = *(const float4*)(sa + 12);
        const float* sb = bRow + k0;
        *(float4*)(bReg + 0)  = *(const float4*)(sb + 0);
        *(float4*)(bReg + 4)  = *(const float4*)(sb + 4);
        *(float4*)(bReg + 8)  = *(const float4*)(sb + 8);
        *(float4*)(bReg + 12) = *(const float4*)(sb + 12);
    };
    // Branchless ternary quant: sign(w)*qs via two selects (v_cmp+v_cndmask),
    // no EXEC-mask control flow in the staging path.
    auto quant = [&](float w) -> float {
        float pos = (w > 0.f) ? qs : 0.f;
        float neg = (w < 0.f) ? qs : 0.f;
        float q   = pos - neg;
        return doq ? q : w;              // uniform select
    };
    auto stage = [&](int p) {
        v8bf pa0, pa1, pb0, pb1;
#pragma unroll
        for (int j = 0; j < 8; ++j) {
            pa0[j] = f2bf(aReg[j]);
            pa1[j] = f2bf(aReg[8 + j]);
            pb0[j] = f2bf(quant(bReg[j]));
            pb1[j] = f2bf(quant(bReg[8 + j]));
        }
        *(v8bf*)&As[p][row][colb]     = pa0;
        *(v8bf*)&As[p][row][colb + 8] = pa1;
        *(v8bf*)&Bs[p][row][colb]     = pb0;
        *(v8bf*)&Bs[p][row][colb + 8] = pb1;
    };

    // prologue: tile 0 into buffer 0
    gload(0);
    stage(0);
    int p = 0;

    for (int k0 = 0; k0 < K; k0 += 32) {
        __syncthreads();                         // buffer p ready for all waves
        const bool hasNext = (k0 + 32) < K;
        if (hasNext) {
            gload(k0 + 32);                      // global loads overlap WMMA below
            if (k0 + 64 < K) {                   // prefetch tile k+2 into caches
                __builtin_prefetch(aRow + k0 + 64, 0, 1);
                __builtin_prefetch(bRow + k0 + 64, 0, 1);
            }
        }

        // ---- load fragments per documented wave32 layouts ----
        v16bf afr[2], bfr[2];
#pragma unroll
        for (int s = 0; s < 2; ++s) {
            const bf16_t* ap = &As[p][wm * 32 + s * 16 + fr][kb8];
            const v8bf lo = *(const v8bf*)(ap);
            const v8bf hi = *(const v8bf*)(ap + 16);
#pragma unroll
            for (int e = 0; e < 8; ++e) { afr[s][e] = lo[e]; afr[s][8 + e] = hi[e]; }
        }
#pragma unroll
        for (int s = 0; s < 2; ++s) {
            const bf16_t* bp = &Bs[p][wn * 32 + s * 16 + fr][kb8];
            const v8bf l0 = *(const v8bf*)(bp);
            const v8bf l1 = *(const v8bf*)(bp + 16);
#pragma unroll
            for (int e = 0; e < 8; ++e) { bfr[s][e] = l0[e]; bfr[s][8 + e] = l1[e]; }
        }

#pragma unroll
        for (int i = 0; i < 2; ++i)
#pragma unroll
            for (int j = 0; j < 2; ++j)
                acc[i][j] = wmma_bf16(afr[i], bfr[j], acc[i][j]);

        if (hasNext) {
            stage(p ^ 1);                        // write other buffer: no conflict
            p ^= 1;
        }
    }

    // ---- epilogue + store (C-layout: vgpr r, lane L -> row r+8*(L>=16), col L&15) ----
    const int nc    = lane & 15;
    const int mhalf = (lane >> 4) * 8;
#pragma unroll
    for (int i = 0; i < 2; ++i) {
#pragma unroll
        for (int j = 0; j < 2; ++j) {
            const int col = n0 + wn * 32 + j * 16 + nc;
#pragma unroll
            for (int r = 0; r < 8; ++r) {
                const int rowm = m0 + wm * 32 + i * 16 + mhalf + r;
                float v = acc[i][j][r];
                if (do_gelu) v = 0.5f * v * (1.f + erff(v * 0.7071067811865475f));
                v *= beta;
                if (resid) v += resid[(size_t)rowm * N + col];
                out[(size_t)rowm * N + col] = v;
            }
        }
    }
}

// ---------------------------------------------------------------------------
// Flash-style causal attention, head_dim = 64.  One wave per 16-query tile.
// grid = (T/16, H, B); q/k/v/y are [B*T, C] with head slice [h*64, h*64+64).
// ---------------------------------------------------------------------------
__global__ void attn_kernel(const float* __restrict__ Q,
                            const float* __restrict__ Kb,
                            const float* __restrict__ V,
                            float* __restrict__ Y,
                            int C, int T) {
    const int lane = threadIdx.x;
    const int qt = blockIdx.x, h = blockIdx.y, b = blockIdx.z;
    const int q0 = qt * 16;
    const size_t base = (size_t)b * T * C + (size_t)h * 64;

    __shared__ bf16_t Pl[16][32];

    const int fr = lane & 15;
    const int hi = lane >> 4;

    // Q fragments (A layout), softmax scale 1/sqrt(64) folded in
    v16bf aQ[2];
    {
        const float* qrow = Q + base + (size_t)(q0 + fr) * C;
#pragma unroll
        for (int c = 0; c < 2; ++c) {
            const int kb = c * 32 + hi * 8;
            const float4 f0 = *(const float4*)(qrow + kb);
            const float4 f1 = *(const float4*)(qrow + kb + 4);
            const float4 f2 = *(const float4*)(qrow + kb + 16);
            const float4 f3 = *(const float4*)(qrow + kb + 20);
            aQ[c][0]  = f2bf(f0.x * 0.125f); aQ[c][1]  = f2bf(f0.y * 0.125f);
            aQ[c][2]  = f2bf(f0.z * 0.125f); aQ[c][3]  = f2bf(f0.w * 0.125f);
            aQ[c][4]  = f2bf(f1.x * 0.125f); aQ[c][5]  = f2bf(f1.y * 0.125f);
            aQ[c][6]  = f2bf(f1.z * 0.125f); aQ[c][7]  = f2bf(f1.w * 0.125f);
            aQ[c][8]  = f2bf(f2.x * 0.125f); aQ[c][9]  = f2bf(f2.y * 0.125f);
            aQ[c][10] = f2bf(f2.z * 0.125f); aQ[c][11] = f2bf(f2.w * 0.125f);
            aQ[c][12] = f2bf(f3.x * 0.125f); aQ[c][13] = f2bf(f3.y * 0.125f);
            aQ[c][14] = f2bf(f3.z * 0.125f); aQ[c][15] = f2bf(f3.w * 0.125f);
        }
    }

    float mstat[8], lstat[8];
#pragma unroll
    for (int r = 0; r < 8; ++r) { mstat[r] = -3.0e38f; lstat[r] = 0.f; }
    v8f oacc[4] = {};

    const int nk2 = (q0 >> 5) + 1;      // 32-key steps (causal)
    for (int kt = 0; kt < nk2; ++kt) {
        const int kbase = kt * 32;

        // ---- S = Q*K^T for two 16-key subtiles ----
        v8f S[2];
#pragma unroll
        for (int s = 0; s < 2; ++s) {
            v16bf bK[2];
            const float* krow = Kb + base + (size_t)(kbase + s * 16 + fr) * C;
#pragma unroll
            for (int c = 0; c < 2; ++c) {
                const int hb = c * 32 + hi * 8;
                const float4 g0 = *(const float4*)(krow + hb);
                const float4 g1 = *(const float4*)(krow + hb + 4);
                const float4 g2 = *(const float4*)(krow + hb + 16);
                const float4 g3 = *(const float4*)(krow + hb + 20);
                bK[c][0]  = f2bf(g0.x); bK[c][1]  = f2bf(g0.y);
                bK[c][2]  = f2bf(g0.z); bK[c][3]  = f2bf(g0.w);
                bK[c][4]  = f2bf(g1.x); bK[c][5]  = f2bf(g1.y);
                bK[c][6]  = f2bf(g1.z); bK[c][7]  = f2bf(g1.w);
                bK[c][8]  = f2bf(g2.x); bK[c][9]  = f2bf(g2.y);
                bK[c][10] = f2bf(g2.z); bK[c][11] = f2bf(g2.w);
                bK[c][12] = f2bf(g3.x); bK[c][13] = f2bf(g3.y);
                bK[c][14] = f2bf(g3.z); bK[c][15] = f2bf(g3.w);
            }
            v8f sacc = {};
            sacc = wmma_bf16(aQ[0], bK[0], sacc);
            sacc = wmma_bf16(aQ[1], bK[1], sacc);
            S[s] = sacc;
        }

        // ---- causal mask + online softmax ----
        float alpha[8];
#pragma unroll
        for (int r = 0; r < 8; ++r) {
            const int qi = q0 + hi * 8 + r;
#pragma unroll
            for (int s = 0; s < 2; ++s) {
                const int key = kbase + s * 16 + fr;
                float v = S[s][r];
                if (key > qi) v = -1e30f;
                S[s][r] = v;
            }
            float vmax = fmaxf(S[0][r], S[1][r]);
            vmax = fmaxf(vmax, __shfl_xor(vmax, 1, 32));
            vmax = fmaxf(vmax, __shfl_xor(vmax, 2, 32));
            vmax = fmaxf(vmax, __shfl_xor(vmax, 4, 32));
            vmax = fmaxf(vmax, __shfl_xor(vmax, 8, 32));
            const float mnew = fmaxf(mstat[r], vmax);
            alpha[r] = expf(mstat[r] - mnew);
            float psum = 0.f;
#pragma unroll
            for (int s = 0; s < 2; ++s) {
                const float p = expf(S[s][r] - mnew);
                S[s][r] = p;
                psum += p;
            }
            psum += __shfl_xor(psum, 1, 32);
            psum += __shfl_xor(psum, 2, 32);
            psum += __shfl_xor(psum, 4, 32);
            psum += __shfl_xor(psum, 8, 32);
            lstat[r] = lstat[r] * alpha[r] + psum;
            mstat[r] = mnew;
        }
#pragma unroll
        for (int t = 0; t < 4; ++t)
#pragma unroll
            for (int r = 0; r < 8; ++r) oacc[t][r] *= alpha[r];

        // ---- restage P (C-layout) -> LDS -> A-fragment layout ----
        __syncthreads();
#pragma unroll
        for (int s = 0; s < 2; ++s)
#pragma unroll
            for (int r = 0; r < 8; ++r)
                Pl[hi * 8 + r][s * 16 + fr] = f2bf(S[s][r]);
        __syncthreads();

        v16bf aP;
        {
            const bf16_t* pp = &Pl[fr][hi * 8];
            const v8bf lo = *(const v8bf*)(pp);
            const v8bf hh = *(const v8bf*)(pp + 16);
#pragma unroll
            for (int e = 0; e < 8; ++e) { aP[e] = lo[e]; aP[8 + e] = hh[e]; }
        }

        // ---- O += P * V  (4 output hd-chunks of 16) ----
#pragma unroll
        for (int t = 0; t < 4; ++t) {
            v16bf bV;
#pragma unroll
            for (int e = 0; e < 16; ++e) {
                const int kr = kbase + hi * 8 + (e < 8 ? e : e + 8);
                bV[e] = f2bf(V[base + (size_t)kr * C + t * 16 + fr]);
            }
            oacc[t] = wmma_bf16(aP, bV, oacc[t]);
        }
    }

    // ---- normalize + store ----
#pragma unroll
    for (int t = 0; t < 4; ++t)
#pragma unroll
        for (int r = 0; r < 8; ++r) {
            const float v = oacc[t][r] / lstat[r];
            Y[base + (size_t)(q0 + hi * 8 + r) * C + t * 16 + fr] = v;
        }
}

// ---------------------------------------------------------------------------
// Host orchestration
// ---------------------------------------------------------------------------
static void launch_gemm(hipStream_t s, const float* act, const float* W,
                        const float* scale, const float* resid, float beta,
                        int gelu, float* out, int M, int N, int K) {
    dim3 grid(N / 64, M / 64), block(128);
    wmma_gemm_kernel<<<grid, block, 0, s>>>(act, W, scale, resid, beta, gelu, out, M, N, K);
}

extern "C" void kernel_launch(void* const* d_in, const int* in_sizes, int n_in,
                              void* d_out, int out_size, void* d_ws, size_t ws_size,
                              hipStream_t stream) {
    (void)in_sizes; (void)n_in; (void)out_size; (void)ws_size;

    const int B = 4, T = 2048, BT = B * T;
    const int FUSE = 448, AD = 384, BD = 320;
    const int AH = 6, BH = 5, AL = 8, BL = 5;
    const int AM = AD * 3, BM = BD * 3;   // 1152, 960
    const int VOCAB = 1024;

    const int*   ids    = (const int*)  d_in[0];
    const float* tok    = (const float*)d_in[1];
    const float* rproj  = (const float*)d_in[2];
    const float* rgain  = (const float*)d_in[3];
    const float* hemb   = (const float*)d_in[4];
    const float* hgain  = (const float*)d_in[5];
    const float* to_a   = (const float*)d_in[6];
    const float* to_b   = (const float*)d_in[7];
    const float* a_n1   = (const float*)d_in[8];
    const float* a_q    = (const float*)d_in[9];
    const float* a_k    = (const float*)d_in[10];
    const float* a_v    = (const float*)d_in[11];
    const float* a_o    = (const float*)d_in[12];
    const float* a_n2   = (const float*)d_in[13];
    const float* a_fc1  = (const float*)d_in[14];
    const float* a_fc2  = (const float*)d_in[15];
    const float* b_n1   = (const float*)d_in[16];
    const float* b_q    = (const float*)d_in[17];
    const float* b_k    = (const float*)d_in[18];
    const float* b_v    = (const float*)d_in[19];
    const float* b_o    = (const float*)d_in[20];
    const float* b_n2   = (const float*)d_in[21];
    const float* b_fc1  = (const float*)d_in[22];
    const float* b_fc2  = (const float*)d_in[23];
    const float* from_a = (const float*)d_in[24];
    const float* from_b = (const float*)d_in[25];
    const float* fnw    = (const float*)d_in[26];

    // ---- workspace carve (~155 MB) ----
    char* cur = (char*)d_ws;
    auto alloc = [&](size_t bytes) -> float* {
        float* p = (float*)cur;
        cur += ((bytes + 255) & ~(size_t)255);
        return p;
    };
    float* x    = alloc((size_t)BT * FUSE * 4);
    float* xa   = alloc((size_t)BT * AD * 4);
    float* xb   = alloc((size_t)BT * BD * 4);
    float* nrm  = alloc((size_t)BT * FUSE * 4);
    float* qb   = alloc((size_t)BT * AD * 4);
    float* kb   = alloc((size_t)BT * AD * 4);
    float* vb   = alloc((size_t)BT * AD * 4);
    float* yb   = alloc((size_t)BT * AD * 4);
    float* h1   = alloc((size_t)BT * AM * 4);
    float* fus  = alloc((size_t)BT * FUSE * 4);
    float* scal = alloc(96 * 4);

    // ---- BitNet scales (78 quantized matrices) ----
    auto sc = [&](const float* src, int n, int slot) {
        absmean_kernel<<<1, 256, 0, stream>>>(src, n, scal + slot);
    };
    for (int i = 0; i < AL; ++i) {
        sc(a_q   + (size_t)i * AD * AD, AD * AD, i * 6 + 0);
        sc(a_k   + (size_t)i * AD * AD, AD * AD, i * 6 + 1);
        sc(a_v   + (size_t)i * AD * AD, AD * AD, i * 6 + 2);
        sc(a_o   + (size_t)i * AD * AD, AD * AD, i * 6 + 3);
        sc(a_fc1 + (size_t)i * AM * AD, AM * AD, i * 6 + 4);
        sc(a_fc2 + (size_t)i * AD * AM, AD * AM, i * 6 + 5);
    }
    for (int i = 0; i < BL; ++i) {
        sc(b_q   + (size_t)i * BD * BD, BD * BD, 48 + i * 6 + 0);
        sc(b_k   + (size_t)i * BD * BD, BD * BD, 48 + i * 6 + 1);
        sc(b_v   + (size_t)i * BD * BD, BD * BD, 48 + i * 6 + 2);
        sc(b_o   + (size_t)i * BD * BD, BD * BD, 48 + i * 6 + 3);
        sc(b_fc1 + (size_t)i * BM * BD, BM * BD, 48 + i * 6 + 4);
        sc(b_fc2 + (size_t)i * BD * BM, BD * BM, 48 + i * 6 + 5);
    }

    // ---- embedding + radial + hash ----
    embed_kernel<<<dim3(BT), dim3(FUSE), 0, stream>>>(ids, tok, rproj, rgain, hemb, hgain, x, T);

    // ---- branch projections ----
    launch_gemm(stream, x, to_a, nullptr, nullptr, 1.f, 0, xa, BT, AD, FUSE);
    launch_gemm(stream, x, to_b, nullptr, nullptr, 1.f, 0, xb, BT, BD, FUSE);

    // ---- branch A ----
    for (int i = 0; i < AL; ++i) {
        const float* Wq = a_q   + (size_t)i * AD * AD;
        const float* Wk = a_k   + (size_t)i * AD * AD;
        const float* Wv = a_v   + (size_t)i * AD * AD;
        const float* Wo = a_o   + (size_t)i * AD * AD;
        const float* W1 = a_fc1 + (size_t)i * AM * AD;
        const float* W2 = a_fc2 + (size_t)i * AD * AM;
        rmsnorm_kernel<<<dim3(BT), dim3(256), 0, stream>>>(xa, a_n1 + (size_t)i * AD, nrm, AD);
        launch_gemm(stream, nrm, Wq, scal + i * 6 + 0, nullptr, 1.f, 0, qb, BT, AD, AD);
        launch_gemm(stream, nrm, Wk, scal + i * 6 + 1, nullptr, 1.f, 0, kb, BT, AD, AD);
        launch_gemm(stream, nrm, Wv, scal + i * 6 + 2, nullptr, 1.f, 0, vb, BT, AD, AD);
        attn_kernel<<<dim3(T / 16, AH, B), dim3(32), 0, stream>>>(qb, kb, vb, yb, AD, T);
        launch_gemm(stream, yb, Wo, scal + i * 6 + 3, xa, 1.f, 0, xa, BT, AD, AD);
        rmsnorm_kernel<<<dim3(BT), dim3(256), 0, stream>>>(xa, a_n2 + (size_t)i * AD, nrm, AD);
        launch_gemm(stream, nrm, W1, scal + i * 6 + 4, nullptr, 1.f, 1, h1, BT, AM, AD);
        launch_gemm(stream, h1, W2, scal + i * 6 + 5, xa, 1.f, 0, xa, BT, AD, AM);
    }

    // ---- branch B ----
    for (int i = 0; i < BL; ++i) {
        const float* Wq = b_q   + (size_t)i * BD * BD;
        const float* Wk = b_k   + (size_t)i * BD * BD;
        const float* Wv = b_v   + (size_t)i * BD * BD;
        const float* Wo = b_o   + (size_t)i * BD * BD;
        const float* W1 = b_fc1 + (size_t)i * BM * BD;
        const float* W2 = b_fc2 + (size_t)i * BD * BM;
        const int s0 = 48 + i * 6;
        rmsnorm_kernel<<<dim3(BT), dim3(256), 0, stream>>>(xb, b_n1 + (size_t)i * BD, nrm, BD);
        launch_gemm(stream, nrm, Wq, scal + s0 + 0, nullptr, 1.f, 0, qb, BT, BD, BD);
        launch_gemm(stream, nrm, Wk, scal + s0 + 1, nullptr, 1.f, 0, kb, BT, BD, BD);
        launch_gemm(stream, nrm, Wv, scal + s0 + 2, nullptr, 1.f, 0, vb, BT, BD, BD);
        attn_kernel<<<dim3(T / 16, BH, B), dim3(32), 0, stream>>>(qb, kb, vb, yb, BD, T);
        launch_gemm(stream, yb, Wo, scal + s0 + 3, xb, 1.f, 0, xb, BT, BD, BD);
        rmsnorm_kernel<<<dim3(BT), dim3(256), 0, stream>>>(xb, b_n2 + (size_t)i * BD, nrm, BD);
        launch_gemm(stream, nrm, W1, scal + s0 + 4, nullptr, 1.f, 1, h1, BT, BM, BD);
        launch_gemm(stream, h1, W2, scal + s0 + 5, xb, 1.f, 0, xb, BT, BD, BM);
    }

    // ---- fuse + final norm + tied LM head ----
    launch_gemm(stream, xa, from_a, nullptr, nullptr, 1.f,   0, fus, BT, FUSE, AD);
    launch_gemm(stream, xb, from_b, nullptr, fus,     0.36f, 0, fus, BT, FUSE, BD);
    rmsnorm_kernel<<<dim3(BT), dim3(256), 0, stream>>>(fus, fnw, nrm, FUSE);
    launch_gemm(stream, nrm, tok, nullptr, nullptr, 1.f, 0, (float*)d_out, BT, VOCAB, FUSE);
}
